// PaiNN_11347303596610
// MI455X (gfx1250) — matrix operand order repaired and checked by reading, not compiled, not run on hardware
//
#include <hip/hip_runtime.h>
#include <hip/hip_bf16.h>

#define NN    8192
#define DEG   16
#define EE    (NN * DEG)
#define FF    128
#define RR    20
#define LL    3
#define CUTD  3.0f
#define PIF   3.14159265358979323846f

typedef __attribute__((ext_vector_type(16))) _Float16 v16h;
typedef __attribute__((ext_vector_type(8)))  float    v8f;

__device__ __forceinline__ float silu_f(float x) {
    // fast path: v_exp_f32 + v_rcp_f32 instead of IEEE div ladder
    return x * __fdividef(1.0f, 1.0f + __expf(-x));
}

// -------------------------------------------------------------------------
// Pack a weight matrix W[K,Nd] (f32, row-major) into f16 WMMA-B fragment
// order: tile (kt,nt) holds, for lane l, 16 halves with element i =
// W[32*kt + 16*(l>=16) + i][16*nt + (l&15)].  One 32B store per lane.
// grid = (K/32)*(Nd/16) blocks x 32 threads.
// -------------------------------------------------------------------------
__global__ __launch_bounds__(32)
void pack_w_kernel(const float* __restrict__ W, v16h* __restrict__ P, int K, int Nd)
{
    const int ntc  = Nd >> 4;
    const int kt   = blockIdx.x / ntc;
    const int nt   = blockIdx.x - kt * ntc;
    const int lane = threadIdx.x;
    const int n    = (nt << 4) + (lane & 15);
    const int kb   = (kt << 5) + ((lane >> 4) << 4);
    v16h r;
#pragma unroll
    for (int i = 0; i < 16; ++i) r[i] = (_Float16)W[(size_t)(kb + i) * Nd + n];
    P[((size_t)blockIdx.x << 5) + lane] = r;
}

// -------------------------------------------------------------------------
// Dense GEMM via v_wmma_f32_16x16x32_f16 with pre-packed f16 B.
// C[M,Nd] = op(A[M,K] @ W[K,Nd] + b).  block = 128 (4 waves); each wave
// computes a 16x32 slab (one A fragment reused for two B tiles).
// grid = (M/16, Nd/128).  K % 32 == 0, Nd % 128 == 0.
// -------------------------------------------------------------------------
__global__ __launch_bounds__(128)
void gemm_wmma_f16(const float* __restrict__ A, const v16h* __restrict__ P,
                   const float* __restrict__ bias, float* __restrict__ C,
                   int M, int K, int Nd, int doBias, int doSilu)
{
    const int wave = threadIdx.x >> 5;
    const int lane = threadIdx.x & 31;
    const int m0   = blockIdx.x << 4;
    const int nt0  = (blockIdx.y << 3) + (wave << 1);   // first of two 16-col tiles
    const int rl   = lane & 15;
    const int hi   = lane >> 4;
    const int ntc  = Nd >> 4;

    // A: lane holds row (m0+rl); K-groups {0..15} (lo) / {16..31} (hi)
    const float* arow = A + (size_t)(m0 + rl) * K + (hi << 4);

    v8f acc0 = {}, acc1 = {};
    for (int k0 = 0, kt = 0; k0 < K; k0 += 32, ++kt) {
        float4 a0 = *(const float4*)(arow + k0 + 0);
        float4 a1 = *(const float4*)(arow + k0 + 4);
        float4 a2 = *(const float4*)(arow + k0 + 8);
        float4 a3 = *(const float4*)(arow + k0 + 12);
        v16h af;
        af[0]  = (_Float16)a0.x; af[1]  = (_Float16)a0.y; af[2]  = (_Float16)a0.z; af[3]  = (_Float16)a0.w;
        af[4]  = (_Float16)a1.x; af[5]  = (_Float16)a1.y; af[6]  = (_Float16)a1.z; af[7]  = (_Float16)a1.w;
        af[8]  = (_Float16)a2.x; af[9]  = (_Float16)a2.y; af[10] = (_Float16)a2.z; af[11] = (_Float16)a2.w;
        af[12] = (_Float16)a3.x; af[13] = (_Float16)a3.y; af[14] = (_Float16)a3.z; af[15] = (_Float16)a3.w;

        const v16h* tb = P + (((size_t)kt * ntc + nt0) << 5) + lane;
        v16h b0 = tb[0];
        v16h b1 = tb[32];
        acc0 = __builtin_amdgcn_wmma_f32_16x16x32_f16(false, af, false, b0,
                                                      (short)0, acc0, false, false);
        acc1 = __builtin_amdgcn_wmma_f32_16x16x32_f16(false, af, false, b1,
                                                      (short)0, acc1, false, false);
    }

    const int n0 = nt0 << 4;
    const int nA = n0 + rl, nB = n0 + 16 + rl;
    const float bvA = doBias ? bias[nA] : 0.0f;
    const float bvB = doBias ? bias[nB] : 0.0f;
#pragma unroll
    for (int r = 0; r < 8; ++r) {
        const size_t mrow = (size_t)(m0 + (hi << 3) + r) * Nd;
        float x = acc0[r] + bvA;
        float y = acc1[r] + bvB;
        if (doSilu) { x = silu_f(x); y = silu_f(y); }
        C[mrow + nA] = x;
        C[mrow + nB] = y;
    }
}

// -------------------------------------------------------------------------
// Init: s = emb[atoms], v = 0, counts = 0.   grid = N, block = 128
// -------------------------------------------------------------------------
__global__ __launch_bounds__(128)
void init_nodes(const int* __restrict__ atoms, const float* __restrict__ emb,
                float* __restrict__ s, float* __restrict__ v0, float* __restrict__ counts)
{
    const int n = blockIdx.x, f = threadIdx.x;
    s[(size_t)n * FF + f] = emb[(size_t)atoms[n] * FF + f];
    v0[(size_t)n * 3 * FF + f]          = 0.0f;
    v0[(size_t)n * 3 * FF + FF + f]     = 0.0f;
    v0[(size_t)n * 3 * FF + 2 * FF + f] = 0.0f;
    if (f == 0) counts[n] = 0.0f;
}

// -------------------------------------------------------------------------
// Per-edge precompute: sinusoidal rbf, cosine cutoff, degree counts.
// -------------------------------------------------------------------------
__global__ void edge_pre(const int* __restrict__ idx_i, const float* __restrict__ rel_dist,
                         float* __restrict__ rbfb, float* __restrict__ cutb,
                         float* __restrict__ counts)
{
    const int e = blockIdx.x * blockDim.x + threadIdx.x;
    if (e >= EE) return;
    const float d = rel_dist[e];
    cutb[e] = 0.5f * (cosf(PIF * d / CUTD) + 1.0f);
    const float inv = 1.0f / d;
#pragma unroll
    for (int r = 0; r < RR; ++r)
        rbfb[(size_t)e * RR + r] = sinf((float)(r + 1) * PIF * d / CUTD) * inv;
    atomicAdd(&counts[idx_i[e]], 1.0f);
}

// -------------------------------------------------------------------------
// Fused message block: per destination node (contiguous 16-edge segment,
// idx_i is sorted -> no atomics).  W = (rbf@rbf_w + b)*cut computed on the
// fly with rbf_w columns cached in registers; ds/dv accumulate in regs.
// v double-buffered (vin -> vout).  grid = N, block = 128.
// -------------------------------------------------------------------------
__global__ __launch_bounds__(128)
void msg_kernel(float* __restrict__ s, const float* __restrict__ vin,
                float* __restrict__ vout, const float* __restrict__ phi,
                const float* __restrict__ rbfb, const float* __restrict__ cutb,
                const int* __restrict__ idx_j, const float* __restrict__ rel_dir,
                const float* __restrict__ rbf_w, const float* __restrict__ rbf_b,
                const float* __restrict__ counts)
{
    const int i = blockIdx.x, f = threadIdx.x;

    float wc0[RR], wc1[RR], wc2[RR];
#pragma unroll
    for (int r = 0; r < RR; ++r) {
        wc0[r] = rbf_w[r * (3 * FF) + f];
        wc1[r] = rbf_w[r * (3 * FF) + FF + f];
        wc2[r] = rbf_w[r * (3 * FF) + 2 * FF + f];
    }
    const float b0 = rbf_b[f], b1 = rbf_b[FF + f], b2 = rbf_b[2 * FF + f];

    float ds = 0.0f, dv0 = 0.0f, dv1 = 0.0f, dv2 = 0.0f;
    for (int e = i * DEG; e < i * DEG + DEG; ++e) {
        const int j = idx_j[e];
        const float ct = cutb[e];
        float w0 = b0, w1 = b1, w2 = b2;
#pragma unroll
        for (int r = 0; r < RR; ++r) {
            const float rb = rbfb[(size_t)e * RR + r];   // broadcast load
            w0 += rb * wc0[r]; w1 += rb * wc1[r]; w2 += rb * wc2[r];
        }
        w0 *= ct; w1 *= ct; w2 *= ct;

        const float pvv = phi[(size_t)j * 3 * FF + f]          * w0;
        const float pss = phi[(size_t)j * 3 * FF + FF + f]     * w1;
        const float pvs = phi[(size_t)j * 3 * FF + 2 * FF + f] * w2;
        ds += pss;

        const float d0 = rel_dir[e * 3 + 0], d1 = rel_dir[e * 3 + 1], d2 = rel_dir[e * 3 + 2];
        dv0 += vin[(size_t)j * 3 * FF + f]          * pvv + pvs * d0;
        dv1 += vin[(size_t)j * 3 * FF + FF + f]     * pvv + pvs * d1;
        dv2 += vin[(size_t)j * 3 * FF + 2 * FF + f] * pvv + pvs * d2;
    }
    const float ic = 1.0f / counts[i];
    s[(size_t)i * FF + f] += ds * ic;
    vout[(size_t)i * 3 * FF + f]          = vin[(size_t)i * 3 * FF + f]          + dv0 * ic;
    vout[(size_t)i * 3 * FF + FF + f]     = vin[(size_t)i * 3 * FF + FF + f]     + dv1 * ic;
    vout[(size_t)i * 3 * FF + 2 * FF + f] = vin[(size_t)i * 3 * FF + 2 * FF + f] + dv2 * ic;
}

// a_in = [ ||Vv||_c , s ]   grid = N, block = 128
__global__ __launch_bounds__(128)
void ain_kernel(const float* __restrict__ Vv, const float* __restrict__ s,
                float* __restrict__ a_in)
{
    const int n = blockIdx.x, f = threadIdx.x;
    const float x = Vv[(size_t)n * 3 * FF + f];
    const float y = Vv[(size_t)n * 3 * FF + FF + f];
    const float z = Vv[(size_t)n * 3 * FF + 2 * FF + f];
    a_in[(size_t)n * 2 * FF + f]      = sqrtf(x * x + y * y + z * z);
    a_in[(size_t)n * 2 * FF + FF + f] = s[(size_t)n * FF + f];
}

// s += a_ss + a_sv * <Uv,Vv>;  v += Uv * a_vv     grid = N, block = 128
__global__ __launch_bounds__(128)
void upd_kernel(float* __restrict__ s, float* __restrict__ v,
                const float* __restrict__ Uv, const float* __restrict__ Vv,
                const float* __restrict__ a)
{
    const int n = blockIdx.x, f = threadIdx.x;
    const size_t b = (size_t)n * 3 * FF;
    const float u0 = Uv[b + f], u1 = Uv[b + FF + f], u2 = Uv[b + 2 * FF + f];
    const float w0 = Vv[b + f], w1 = Vv[b + FF + f], w2 = Vv[b + 2 * FF + f];
    const float a_vv = a[b + f];
    const float a_sv = a[b + FF + f];
    const float a_ss = a[b + 2 * FF + f];
    const float dot = u0 * w0 + u1 * w1 + u2 * w2;
    s[(size_t)n * FF + f] += a_ss + a_sv * dot;
    v[b + f]          += u0 * a_vv;
    v[b + FF + f]     += u1 * a_vv;
    v[b + 2 * FF + f] += u2 * a_vv;
}

// d_out = [ s (N,F) , v transposed to (N,F,3) ]
__global__ __launch_bounds__(128)
void out_kernel(const float* __restrict__ s, const float* __restrict__ v,
                float* __restrict__ out)
{
    const int n = blockIdx.x, f = threadIdx.x;
    out[(size_t)n * FF + f] = s[(size_t)n * FF + f];
    float* ov = out + (size_t)NN * FF;
#pragma unroll
    for (int c = 0; c < 3; ++c)
        ov[(size_t)n * 3 * FF + (size_t)f * 3 + c] = v[(size_t)n * 3 * FF + (size_t)c * FF + f];
}

// -------------------------------------------------------------------------
extern "C" void kernel_launch(void* const* d_in, const int* in_sizes, int n_in,
                              void* d_out, int out_size, void* d_ws, size_t ws_size,
                              hipStream_t stream)
{
    (void)in_sizes; (void)n_in; (void)out_size; (void)ws_size;
    const int*   atoms    = (const int*)d_in[0];
    const int*   idx_i    = (const int*)d_in[1];
    const int*   idx_j    = (const int*)d_in[2];
    const float* rel_dir  = (const float*)d_in[3];
    const float* rel_dist = (const float*)d_in[4];
    const float* emb      = (const float*)d_in[5];
    const float* msg_w1   = (const float*)d_in[6];
    const float* msg_b1   = (const float*)d_in[7];
    const float* msg_w2   = (const float*)d_in[8];
    const float* msg_b2   = (const float*)d_in[9];
    const float* rbf_w    = (const float*)d_in[10];
    const float* rbf_b    = (const float*)d_in[11];
    const float* Uw       = (const float*)d_in[12];
    const float* Vw       = (const float*)d_in[13];
    const float* upd_w1   = (const float*)d_in[14];
    const float* upd_b1   = (const float*)d_in[15];
    const float* upd_w2   = (const float*)d_in[16];
    const float* upd_b2   = (const float*)d_in[17];

    float* ws = (float*)d_ws;
    size_t o = 0;
    float* s      = ws + o; o += (size_t)NN * FF;
    float* v0     = ws + o; o += (size_t)NN * 3 * FF;
    float* v1     = ws + o; o += (size_t)NN * 3 * FF;
    float* phi    = ws + o; o += (size_t)NN * 3 * FF;   // reused as 'a' buffer
    float* H      = ws + o; o += (size_t)NN * FF;
    float* Uv     = ws + o; o += (size_t)NN * 3 * FF;
    float* Vv     = ws + o; o += (size_t)NN * 3 * FF;
    float* a_in   = ws + o; o += (size_t)NN * 2 * FF;
    float* rbfb   = ws + o; o += (size_t)EE * RR;
    float* cutb   = ws + o; o += (size_t)EE;
    float* counts = ws + o; o += (size_t)NN;

    // packed f16 weight tiles (WMMA-B fragment order), per layer
    const size_t S1 = (size_t)FF * FF;          // msg_w1 / Uw / Vw
    const size_t S2 = (size_t)FF * 3 * FF;      // msg_w2 / upd_w2
    const size_t SP = (size_t)2 * FF * FF;      // upd_w1
    v16h* pw = (v16h*)(ws + o);                 // halves region
    _Float16* ph = (_Float16*)pw;
    _Float16* p_m1[LL], *p_m2[LL], *p_U[LL], *p_V[LL], *p_u1[LL], *p_u2[LL];
    size_t ho = 0;
    for (int l = 0; l < LL; ++l) {
        p_m1[l] = ph + ho; ho += S1;
        p_m2[l] = ph + ho; ho += S2;
        p_U[l]  = ph + ho; ho += S1;
        p_V[l]  = ph + ho; ho += S1;
        p_u1[l] = ph + ho; ho += SP;
        p_u2[l] = ph + ho; ho += S2;
    }

    init_nodes<<<NN, 128, 0, stream>>>(atoms, emb, s, v0, counts);
    edge_pre<<<(EE + 255) / 256, 256, 0, stream>>>(idx_i, rel_dist, rbfb, cutb, counts);

    for (int l = 0; l < LL; ++l) {
        pack_w_kernel<<<(FF / 32) * (FF / 16), 32, 0, stream>>>(msg_w1 + l * S1, (v16h*)p_m1[l], FF, FF);
        pack_w_kernel<<<(FF / 32) * (3 * FF / 16), 32, 0, stream>>>(msg_w2 + l * S2, (v16h*)p_m2[l], FF, 3 * FF);
        pack_w_kernel<<<(FF / 32) * (FF / 16), 32, 0, stream>>>(Uw + l * S1, (v16h*)p_U[l], FF, FF);
        pack_w_kernel<<<(FF / 32) * (FF / 16), 32, 0, stream>>>(Vw + l * S1, (v16h*)p_V[l], FF, FF);
        pack_w_kernel<<<(2 * FF / 32) * (FF / 16), 32, 0, stream>>>(upd_w1 + l * SP, (v16h*)p_u1[l], 2 * FF, FF);
        pack_w_kernel<<<(FF / 32) * (3 * FF / 16), 32, 0, stream>>>(upd_w2 + l * S2, (v16h*)p_u2[l], FF, 3 * FF);
    }

    float* vcur = v0;
    float* vnxt = v1;
    for (int l = 0; l < LL; ++l) {
        const size_t oB1 = (size_t)l * FF;
        const size_t oB3 = (size_t)l * 3 * FF;
        // message MLP: H = silu(s@W1+b1); phi = H@W2+b2
        gemm_wmma_f16<<<dim3(NN / 16, 1), 128, 0, stream>>>(s, (v16h*)p_m1[l], msg_b1 + oB1, H,
                                                            NN, FF, FF, 1, 1);
        gemm_wmma_f16<<<dim3(NN / 16, 3), 128, 0, stream>>>(H, (v16h*)p_m2[l], msg_b2 + oB3, phi,
                                                            NN, FF, 3 * FF, 1, 0);
        // fused edge/aggregation (reads vcur, writes vnxt; s in place)
        msg_kernel<<<NN, 128, 0, stream>>>(s, vcur, vnxt, phi, rbfb, cutb, idx_j, rel_dir,
                                           rbf_w + (size_t)l * RR * 3 * FF, rbf_b + oB3, counts);
        float* t = vcur; vcur = vnxt; vnxt = t;
        // Uv/Vv: v viewed as [3N, F] rows
        gemm_wmma_f16<<<dim3(3 * NN / 16, 1), 128, 0, stream>>>(vcur, (v16h*)p_U[l], nullptr, Uv,
                                                                3 * NN, FF, FF, 0, 0);
        gemm_wmma_f16<<<dim3(3 * NN / 16, 1), 128, 0, stream>>>(vcur, (v16h*)p_V[l], nullptr, Vv,
                                                                3 * NN, FF, FF, 0, 0);
        ain_kernel<<<NN, 128, 0, stream>>>(Vv, s, a_in);
        // update MLP: H = silu(a_in@uw1+ub1); a = H@uw2+ub2  (a aliases phi)
        gemm_wmma_f16<<<dim3(NN / 16, 1), 128, 0, stream>>>(a_in, (v16h*)p_u1[l], upd_b1 + oB1, H,
                                                            NN, 2 * FF, FF, 1, 1);
        gemm_wmma_f16<<<dim3(NN / 16, 3), 128, 0, stream>>>(H, (v16h*)p_u2[l], upd_b2 + oB3, phi,
                                                            NN, FF, 3 * FF, 1, 0);
        upd_kernel<<<NN, 128, 0, stream>>>(s, vcur, Uv, Vv, phi);
    }
    out_kernel<<<NN, 128, 0, stream>>>(s, vcur, (float*)d_out);
}
